// NodeFeatByVN_7885559956062
// MI455X (gfx1250) — matrix operand order repaired and checked by reading, not compiled, not run on hardware
//
#include <hip/hip_runtime.h>
#include <math.h>

typedef __attribute__((ext_vector_type(16))) _Float16 v16h;
typedef __attribute__((ext_vector_type(8)))  _Float16 v8h;
typedef __attribute__((ext_vector_type(8)))  float    v8f;
typedef __attribute__((ext_vector_type(4)))  float    v4f;

// fast reciprocal: single v_rcp_f32 (precision fine next to f16 WMMA inputs)
__device__ __forceinline__ float fast_rcp(float x) {
    return __builtin_amdgcn_rcpf(x);
}

// ---------------------------------------------------------------------------
// Phase 1: per-GROUP MLP (output depends only on batch group, node_pos cancels)
// One wave (32 lanes) handles 16 groups via WMMA f16 (f32 accumulate).
// ---------------------------------------------------------------------------
__global__ __launch_bounds__(256) void group_mlp_kernel(
    const float* __restrict__ vn_pos,   // [G, 12]
    const float* __restrict__ W1,       // [64, 16]
    const float* __restrict__ b1,       // [64]
    const float* __restrict__ W2,       // [64, 64]
    const float* __restrict__ b2,       // [64]
    float* __restrict__ grp_out,        // [G, 64]
    int G)
{
    __shared__ __align__(16) _Float16 hbuf[8][16][64];  // per-wave 16x64 h tile (f16)

    const int lane  = threadIdx.x & 31;
    const int wid   = threadIdx.x >> 5;
    const int tile  = blockIdx.x * 8 + wid;
    const int gbase = tile * 16;
    const int row   = lane & 15;           // group-in-tile / column-in-tile index
    const int mbase = (lane < 16) ? 0 : 8; // C/D layout M offset for this half-wave

    int gid = gbase + row;
    if (gid >= G) gid = G - 1;

    // ---- pairwise distances among the 4 virtual-node positions of this group
    const float* vp = vn_pos + (size_t)gid * 12;
    float wx[4], wy[4], wz[4];
#pragma unroll
    for (int a = 0; a < 4; ++a) { wx[a] = vp[3*a]; wy[a] = vp[3*a+1]; wz[a] = vp[3*a+2]; }

    float dm[4][4];
    float ss = 0.0f;
#pragma unroll
    for (int a = 0; a < 4; ++a) dm[a][a] = 0.0f;
#pragma unroll
    for (int a = 0; a < 4; ++a)
#pragma unroll
        for (int b = a + 1; b < 4; ++b) {
            float dx = wx[b] - wx[a], dy = wy[b] - wy[a], dz = wz[b] - wz[a];
            float d2 = dx*dx + dy*dy + dz*dz;
            float d  = sqrtf(d2);
            dm[a][b] = d; dm[b][a] = d;
            ss += d2;
        }
    // ||info_scalar||_2 over all 16 entries = sqrt(2 * sum_{a<b} d^2)
    const float inv = fast_rcp(sqrtf(2.0f * ss) + 0.001f);

    // ---- A fragment (16x32 f16, K=0..15 real, K=16..31 zero)
    // lanes 0-15: elems 0..7 = K 0..7 ; lanes 16-31: elems 0..7 = K 8..15
    v16h afrag;
    {
        const int kb = (lane < 16) ? 0 : 8;
#pragma unroll
        for (int j = 0; j < 8; ++j) {
            const int k = kb + j;
            afrag[j]     = (_Float16)(dm[k >> 2][k & 3] * inv);
            afrag[8 + j] = (_Float16)0.0f;
        }
    }

    // ---- B1 fragments (W1^T tiles, 32x16: lanes 0-15 elems 0..7 = K0..7, lanes 16-31 elems 0..7 = K8..15)
    v16h  b1f[4];
    float b1v[4], b2v[4];
#pragma unroll
    for (int t = 0; t < 4; ++t) {
        const int col = 16 * t + row;
        b1v[t] = b1[col];
        b2v[t] = b2[col];
        {
            const float* wr = W1 + col * 16 + ((lane < 16) ? 0 : 8);   // W1^T[k][col] = W1[col][k]
#pragma unroll
            for (int j = 0; j < 8; ++j) { b1f[t][j] = (_Float16)wr[j]; b1f[t][8 + j] = (_Float16)0.0f; } // K=16..31 pad
        }
    }

    // ---- B2 fragments (W2^T, K=64 -> two 32x16 chunks per 16-col tile)
    v16h b2f[4][2];
#pragma unroll
    for (int t = 0; t < 4; ++t)
#pragma unroll
        for (int c = 0; c < 2; ++c) {
            const float* wr = W2 + (size_t)(16*t + row) * 64 + 32*c + ((lane < 16) ? 0 : 16);
#pragma unroll
            for (int j = 0; j < 16; ++j) b2f[t][c][j] = (_Float16)wr[j];
        }

    // ---- GEMM1 (16x16x32, K padded) + bias + silu -> LDS (f16, C/D layout)
#pragma unroll
    for (int t = 0; t < 4; ++t) {
        v8f acc = {};
        acc = __builtin_amdgcn_wmma_f32_16x16x32_f16(
            false, afrag, false, b1f[t], (short)0, acc, false, false);
#pragma unroll
        for (int r = 0; r < 8; ++r) {
            float x  = acc[r] + b1v[t];
            float hx = x * fast_rcp(1.0f + __expf(-x));   // silu, v_rcp_f32
            hbuf[wid][mbase + r][16 * t + row] = (_Float16)hx;
        }
    }

    // ---- re-read h in A layout (16x32 per K-chunk) via 16B LDS vector loads
    v16h ha[2];
    {
        const v8h* hp = (const v8h*)&hbuf[wid][row][0]; // 8 x (8 halves) per row
        const int  o  = (lane < 16) ? 0 : 1;
#pragma unroll
        for (int c = 0; c < 2; ++c) {
            v8h lo = hp[4 * c + 2 * o];       // K = 32c + {0|16} .. +7
            v8h hi = hp[4 * c + 2 * o + 1];   // K = 32c + 8 + {0|16} .. +7
#pragma unroll
            for (int j = 0; j < 8; ++j) { ha[c][j] = lo[j]; ha[c][8 + j] = hi[j]; }
        }
    }

    // ---- GEMM2: two chained K=32 WMMAs per 16-col tile, + bias, store table
#pragma unroll
    for (int t = 0; t < 4; ++t) {
        v8f acc = {};
        acc = __builtin_amdgcn_wmma_f32_16x16x32_f16(
            false, ha[0], false, b2f[t][0], (short)0, acc, false, false);
        acc = __builtin_amdgcn_wmma_f32_16x16x32_f16(
            false, ha[1], false, b2f[t][1], (short)0, acc, false, false);
#pragma unroll
        for (int r = 0; r < 8; ++r) {
            const int gm = gbase + mbase + r;
            if (gm < G)
                grp_out[(size_t)gm * 64 + 16 * t + row] = acc[r] + b2v[t];
        }
    }
}

// ---------------------------------------------------------------------------
// Phase 2: gather [N,64] = table[batch[i]]. One thread per float4 of output:
// perfectly coalesced 128-bit non-temporal streaming stores (store-BW bound).
// ---------------------------------------------------------------------------
__global__ __launch_bounds__(256) void gather_kernel(
    const v4f* __restrict__ grp,          // [G*16] float4
    const long long* __restrict__ batch,  // [N] int64
    v4f* __restrict__ out,                // [N*16] float4
    long long total4)
{
    long long tid = (long long)blockIdx.x * blockDim.x + threadIdx.x;
    if (tid >= total4) return;
    const int node = (int)(tid >> 4);
    const int j    = (int)(tid & 15);
    const long long g = batch[node];      // batch is sorted -> cache friendly
    v4f v = grp[(size_t)g * 16 + j];      // 2MB table, L2-resident
    __builtin_nontemporal_store(v, out + tid);  // write-once stream
}

// ---------------------------------------------------------------------------
extern "C" void kernel_launch(void* const* d_in, const int* in_sizes, int n_in,
                              void* d_out, int out_size, void* d_ws, size_t ws_size,
                              hipStream_t stream)
{
    // inputs: 0 node_feat(unused), 1 node_pos(cancels out), 2 vn_pos, 3 batch,
    //         4 W1, 5 b1, 6 W2, 7 b2
    const float*     vn_pos = (const float*)d_in[2];
    const long long* batch  = (const long long*)d_in[3];
    const float*     W1     = (const float*)d_in[4];
    const float*     b1     = (const float*)d_in[5];
    const float*     W2     = (const float*)d_in[6];
    const float*     b2     = (const float*)d_in[7];
    float*           out    = (float*)d_out;

    const int N = in_sizes[3];         // number of nodes
    const int G = in_sizes[2] / 12;    // number of groups (vn_pos = [G, 3*4])

    float* grp = (float*)d_ws;         // [G, 64] group output table (G*64*4 bytes)

    const int tiles   = (G + 15) / 16;
    const int blocks1 = (tiles + 7) / 8;   // 8 waves (tiles) per 256-thread block
    group_mlp_kernel<<<blocks1, 256, 0, stream>>>(vn_pos, W1, b1, W2, b2, grp, G);

    const long long total4  = (long long)N * 16;
    const int       blocks2 = (int)((total4 + 255) / 256);
    gather_kernel<<<blocks2, 256, 0, stream>>>((const v4f*)grp, batch, (v4f*)out, total4);
}
